// LshSimmatModule_74466142978331
// MI455X (gfx1250) — 8-bit matrix operand order repaired and checked by reading, compiled, not run on hardware
//
#include <hip/hip_runtime.h>
#include <cmath>

#define L_DIM  2
#define BATCH  32
#define A_DIM  64
#define B_DIM  1024
#define D_DIM  128
#define NBITS  1024

typedef __attribute__((ext_vector_type(16))) __bf16 v16bf;
typedef __attribute__((ext_vector_type(8)))  float  v8f;
typedef __attribute__((ext_vector_type(8)))  int    v8i;
typedef __attribute__((ext_vector_type(4)))  int    v4i;
typedef __attribute__((ext_vector_type(2)))  int    v2i;
typedef __attribute__((ext_vector_type(4)))  float  v4f;

__global__ void __launch_bounds__(256) lsh_cvt_r_kernel(
    const float* __restrict__ r, __bf16* __restrict__ rb, int n)
{
    int i = blockIdx.x * blockDim.x + threadIdx.x;
    if (i < n) rb[i] = (__bf16)r[i];
}

__global__ void __launch_bounds__(256) lsh_hash_kernel(
    const float* __restrict__ emb, const __bf16* __restrict__ rb,
    signed char* __restrict__ codes)
{
    const int wave = threadIdx.x >> 5;
    const int lane = threadIdx.x & 31;
    const int half = lane >> 4;
    const int l16  = lane & 15;

    const float* arow = emb + (size_t)(blockIdx.x * 16 + l16) * D_DIM;

    v16bf afrag[4];
#pragma unroll
    for (int ks = 0; ks < 4; ++ks) {
        const int kb = ks * 32;
#pragma unroll
        for (int g = 0; g < 2; ++g) {
            const float* p = arow + kb + g * 16 + half * 8;
            v4f x0 = *(const v4f*)(p);
            v4f x1 = *(const v4f*)(p + 4);
#pragma unroll
            for (int e = 0; e < 4; ++e) {
                afrag[ks][g * 8 + e]     = (__bf16)x0[e];
                afrag[ks][g * 8 + 4 + e] = (__bf16)x1[e];
            }
        }
    }

#pragma unroll 1
    for (int j = 0; j < 8; ++j) {
        const int nt = wave * 8 + j;
        const __bf16* brow = rb + (size_t)(nt * 16 + l16) * D_DIM;
        v8f acc = {};
#pragma unroll
        for (int ks = 0; ks < 4; ++ks) {
            union { v16bf v; v4i q[2]; } b;
            const __bf16* p = brow + ks * 32 + half * 8;
            b.q[0] = *(const v4i*)(p);
            b.q[1] = *(const v4i*)(p + 16);
            acc = __builtin_amdgcn_wmma_f32_16x16x32_bf16(
                false, afrag[ks], false, b.v, (short)0, acc, false, false);
        }
        signed char* crow =
            codes + (size_t)(blockIdx.x * 16 + half * 8) * NBITS + nt * 16 + l16;
#pragma unroll
        for (int i = 0; i < 8; ++i)
            crow[(size_t)i * NBITS] = (acc[i] > 0.0f) ? (signed char)1 : (signed char)-1;
    }
}

__global__ void __launch_bounds__(128) lsh_dot_kernel(
    const signed char* __restrict__ qc, const signed char* __restrict__ dc,
    const int* __restrict__ qt, const int* __restrict__ dt,
    float* __restrict__ out, int l)
{
    const int bat  = blockIdx.x;
    const int wave = threadIdx.x >> 5;
    const int lane = threadIdx.x & 31;
    const int half = lane >> 4;
    const int l16  = lane & 15;

    const signed char* arow = qc + ((size_t)bat * A_DIM + wave * 16 + l16) * NBITS;
    v8i afrag[16];
#pragma unroll
    for (int ks = 0; ks < 16; ++ks) {
        union { v8i v; v2i d[4]; } t;
#pragma unroll
        for (int g = 0; g < 4; ++g)
            t.d[g] = *(const v2i*)(arow + ks * 64 + (g >> 1) * 32 + half * 16 + (g & 1) * 8);
        afrag[ks] = t.v;
    }

    float qm[8];
#pragma unroll
    for (int i = 0; i < 8; ++i)
        qm[i] = (qt[bat * A_DIM + wave * 16 + half * 8 + i] == 0) ? 0.0f : 1.0f;

    const signed char* bbase = dc + (size_t)bat * B_DIM * NBITS;
    float* obase = out + ((size_t)(bat * L_DIM + l) * A_DIM + wave * 16) * B_DIM;

#pragma unroll 1
    for (int j = 0; j < 16; ++j) {
        const int nt = blockIdx.y * 16 + j;
        const signed char* brow = bbase + (size_t)(nt * 16 + l16) * NBITS;
        v8i acc = {};
#pragma unroll
        for (int ks = 0; ks < 16; ++ks) {
            union { v8i v; v4i q[2]; } b;
            const signed char* p = brow + ks * 64 + half * 16;
            b.q[0] = *(const v4i*)(p);
            b.q[1] = *(const v4i*)(p + 32);
            acc = __builtin_amdgcn_wmma_i32_16x16x64_iu8(
                true, afrag[ks], true, b.v, acc, false, false);
        }
        const float dm = (dt[bat * B_DIM + nt * 16 + l16] == 0) ? 0.0f : 1.0f;
#pragma unroll
        for (int i = 0; i < 8; ++i) {
            float sim = __sinf((float)acc[i] * (float)(M_PI / 2048.0)) * qm[i] * dm;
            obase[(size_t)(half * 8 + i) * B_DIM + nt * 16 + l16] = sim;
        }
    }
}

extern "C" void kernel_launch(void* const* d_in, const int* in_sizes, int n_in,
                              void* d_out, int out_size, void* d_ws, size_t ws_size,
                              hipStream_t stream)
{
    const float* qe = (const float*)d_in[0];
    const float* de = (const float*)d_in[1];
    const int*   qt = (const int*)d_in[2];
    const int*   dt = (const int*)d_in[3];
    const float* r  = (const float*)d_in[4];

    char* ws = (char*)d_ws;
    __bf16*      rb = (__bf16*)ws;
    signed char* qc = (signed char*)(ws + (size_t)(256 << 10));
    signed char* dc = (signed char*)(ws + (size_t)(256 << 10) + (size_t)(2 << 20));

    lsh_cvt_r_kernel<<<(NBITS * D_DIM) / 256, 256, 0, stream>>>(r, rb, NBITS * D_DIM);

    for (int l = 0; l < L_DIM; ++l) {
        const float* qel = qe + (size_t)l * BATCH * A_DIM * D_DIM;
        const float* del = de + (size_t)l * BATCH * B_DIM * D_DIM;
        lsh_hash_kernel<<<(BATCH * A_DIM) / 16, 256, 0, stream>>>(qel, rb, qc);
        lsh_hash_kernel<<<(BATCH * B_DIM) / 16, 256, 0, stream>>>(del, rb, dc);
        lsh_dot_kernel<<<dim3(BATCH, 4), 128, 0, stream>>>(qc, dc, qt, dt,
                                                           (float*)d_out, l);
    }
}
